// QCONV2_35880156791517
// MI455X (gfx1250) — matrix operand order repaired and checked by reading, not compiled, not run on hardware
//
#include <hip/hip_runtime.h>
#include <hip/hip_bf16.h>

// Problem constants (from reference): x is (4,4,32,32) f32, weights (2,9) f32,
// out is (4,16,32,32) f32. N = 16384 pixel-rows, 9 qubits -> 512 amplitudes.

typedef _Float16 v16h __attribute__((ext_vector_type(16)));
typedef float    v8f  __attribute__((ext_vector_type(8)));
typedef unsigned int u32x4 __attribute__((ext_vector_type(4)));
typedef int          i32x4 __attribute__((ext_vector_type(4)));
typedef int          i32x8 __attribute__((ext_vector_type(8)));

#if defined(__has_builtin)
#if __has_builtin(__builtin_amdgcn_tensor_load_to_lds) && \
    __has_builtin(__builtin_amdgcn_s_wait_tensorcnt)
#define HAVE_TDM 1
#endif
#endif

// ---------------------------------------------------------------------------
// Kernel 1: build B' = realified U^T in WMMA-friendly transposed layout.
// Block j simulates the 2 BasicEntangler layers on basis state |j> (512 complex
// amps in LDS), giving column j of U. Writes Bt[col][k] (f16, 1024x1024) with:
//   Bt[2k  ][2j] =  Re U[k,j]   Bt[2k  ][2j+1] = -Im U[k,j]
//   Bt[2k+1][2j] =  Im U[k,j]   Bt[2k+1][2j+1] =  Re U[k,j]
// so that C[n,2k]=Re(U psi)_k, C[n,2k+1]=Im(U psi)_k for A=[Re psi|Im psi].
// ---------------------------------------------------------------------------
__global__ void build_U(const float* __restrict__ w, _Float16* __restrict__ Bt) {
    __shared__ float2 st[512];
    const int j   = blockIdx.x;
    const int tid = threadIdx.x;            // 64 threads
    for (int i = tid; i < 512; i += 64) st[i] = make_float2(i == j ? 1.f : 0.f, 0.f);
    __syncthreads();
    for (int l = 0; l < 2; ++l) {
        // RX(w[l][q]) on every wire (qubit q lives at bit 8-q)
        for (int q = 0; q < 9; ++q) {
            const float th = 0.5f * w[l * 9 + q];
            const float cc = cosf(th), ss = sinf(th);
            const int   m  = 1 << (8 - q);
            for (int p = tid; p < 256; p += 64) {
                const int i0 = ((p & ~(m - 1)) << 1) | (p & (m - 1));
                const int i1 = i0 | m;
                const float2 a = st[i0], b = st[i1];
                st[i0] = make_float2(cc * a.x + ss * b.y, cc * a.y - ss * b.x);
                st[i1] = make_float2(ss * a.y + cc * b.x, -ss * a.x + cc * b.y);
            }
            __syncthreads();
        }
        // CNOT ring: control q, target (q+1)%9
        for (int q = 0; q < 9; ++q) {
            const int mc = 1 << (8 - q), mt = 1 << (8 - ((q + 1) % 9));
            for (int i = tid; i < 512; i += 64) {
                if ((i & mc) && !(i & mt)) {
                    const float2 t = st[i];
                    st[i] = st[i | mt];
                    st[i | mt] = t;
                }
            }
            __syncthreads();
        }
    }
    for (int n = tid; n < 1024; n += 64) {
        const float2 v = st[n >> 1];
        const float b0 = (n & 1) ? v.y : v.x;    // Bt[n][2j]
        const float b1 = (n & 1) ? v.x : -v.y;   // Bt[n][2j+1]
        Bt[(size_t)n * 1024 + 2 * j]     = (_Float16)b0;
        Bt[(size_t)n * 1024 + 2 * j + 1] = (_Float16)b1;
    }
}

// ---------------------------------------------------------------------------
// Kernel 2: f16 WMMA GEMM with on-the-fly A generation + signed |.|^2 epilogue.
// Block: 1024 threads = 32 wave32s, M=128 rows. Wave (rg=w>>2, cg=w&3) owns a
// 16-row x 128-col tile of the current 512-col half. B chunks (512 rows x 64 B,
// row stride 2 KB) are staged into LDS by the Tensor Data Mover (one
// tensor_load_to_lds per k-chunk, issued by wave 0, completion via TENSORcnt);
// A fragments come from the rank-1 structure of the product state:
// psi[j] = r_hi(j>>3) * r_lo(j&7) * (-i)^popc(j)  -> ~30 VALU per fragment.
// ---------------------------------------------------------------------------
__global__ void __launch_bounds__(1024)
qgemm(const float* __restrict__ x, const _Float16* __restrict__ Bt,
      float* __restrict__ zrow) {
    __shared__ _Float16 Blds[512][32];   // 32 KB: one K-chunk of half the cols
    __shared__ float    red[4][128];     // cross-wave row partials

    const int tid   = threadIdx.x;
    const int lane  = tid & 31;
    const int wave  = tid >> 5;
    const int khalf = lane >> 4;         // 0: lanes 0-15, 1: lanes 16-31
    const int lsub  = lane & 15;
    const int rg    = wave >> 2;         // row group 0..7
    const int cg    = wave & 3;          // col group 0..3 (within 512-col half)
    const int n0    = blockIdx.x * 128;
    const int n     = n0 + rg * 16 + lsub;   // this lane's A-row (pixel index)

    // decode pixel -> 9 patch angles -> per-lane (cos,sin) of a/2
    const int bimg = n >> 12, ch = (n >> 10) & 3, hh = (n >> 5) & 31, ww = n & 31;
    const float* xb = x + (size_t)(bimg * 4 + ch) * 1024;
    float cq[6], sq[6], rlo[8];
    {
        float cc[9], ss[9];
#pragma unroll
        for (int q = 0; q < 9; ++q) {
            const int di = q / 3, dj = q % 3;
            const int yy = hh + di - 1, zz = ww + dj - 1;
            float a = 0.f;
            if ((unsigned)yy < 32u && (unsigned)zz < 32u) a = xb[yy * 32 + zz];
            cc[q] = cosf(0.5f * a);
            ss[q] = sinf(0.5f * a);
        }
#pragma unroll
        for (int q = 0; q < 6; ++q) { cq[q] = cc[q]; sq[q] = ss[q]; }
#pragma unroll
        for (int v = 0; v < 8; ++v)          // qubits 6,7,8 -> j bits 2,1,0
            rlo[v] = ((v & 4) ? ss[6] : cc[6]) * ((v & 2) ? ss[7] : cc[7]) *
                     ((v & 1) ? ss[8] : cc[8]);
    }

    float zacc[8];
#pragma unroll
    for (int v = 0; v < 8; ++v) zacc[v] = 0.f;

    const v8f zf = {0.f, 0.f, 0.f, 0.f, 0.f, 0.f, 0.f, 0.f};
#if HAVE_TDM
    // Static LDS byte offset of Blds (generic LDS pointer, low 32 bits = offset)
    const unsigned lds_base = (unsigned)(size_t)&Blds[0][0];
#endif

    for (int half = 0; half < 2; ++half) {
        v8f acc[8];
#pragma unroll
        for (int t = 0; t < 8; ++t) acc[t] = zf;

        for (int kc = 0; kc < 32; ++kc) {
            __syncthreads();
#if HAVE_TDM
            if (wave == 0) {
                // D# for a 2-D tile: 32 x 512 f16 elements, row stride 1024 elems.
                const unsigned long long gaddr =
                    (unsigned long long)(const void*)Bt +
                    ((unsigned long long)(half * 512) * 1024ull +
                     (unsigned long long)kc * 32ull) * 2ull;
                const u32x4 g0 = {
                    1u,                                    // count=1, user desc
                    lds_base,                              // lds_addr (bytes)
                    (unsigned)(gaddr & 0xffffffffull),     // global_addr[31:0]
                    (unsigned)(gaddr >> 32) | (2u << 30)   // [56:32] | type=2
                };
                const i32x8 g1 = {
                    (int)(1u << 16),          // data_size=1 (2 bytes/elem)
                    (int)(32u << 16),         // tensor_dim0 = 32 (bits 79:48 lo)
                    (int)(512u << 16),        // tensor_dim1 = 512 (bits 111:80 lo)
                    (int)(32u << 16),         // tile_dim0 = 32 (bits 127:112)
                    (int)512,                 // tile_dim1 = 512 (bits 143:128)
                    (int)1024,                // tensor_dim0_stride = 1024 elems
                    0, 0
                };
                const i32x4 gz4 = {0, 0, 0, 0};
#if __clang_major__ >= 23
                const i32x8 gz8 = {0, 0, 0, 0, 0, 0, 0, 0};
                __builtin_amdgcn_tensor_load_to_lds(g0, g1, gz4, gz4, gz8, 0);
#else
                __builtin_amdgcn_tensor_load_to_lds(g0, g1, gz4, gz4, 0);
#endif
                __builtin_amdgcn_s_wait_tensorcnt(0);
            }
#else
            {   // fallback: cooperative copy of the B chunk
                const int c = tid >> 1, seg = tid & 1;
                const _Float16* src =
                    Bt + (size_t)(half * 512 + c) * 1024 + kc * 32 + seg * 16;
                *(v16h*)&Blds[c][seg * 16] = *(const v16h*)src;
                if (kc + 1 < 32) __builtin_prefetch(src + 32, 0, 1);
            }
#endif
            __syncthreads();

            // --- generate A fragment (16x32 f16 layout per ISA 7.12.2) ---
            // frag K(e) = (e&7) + 8*khalf + 16*(e>=8);  j = kc*16 + (K>>1)
            float rcom = 1.f;
#pragma unroll
            for (int q = 0; q < 5; ++q)
                rcom *= ((kc >> (4 - q)) & 1) ? sq[q] : cq[q];
            const int   pcom = __popc(kc);
            const float rh0 = rcom * cq[5], rh1 = rcom * sq[5];   // u = 2kc+{0,1}
            v16h af;
#pragma unroll
            for (int e = 0; e < 16; ++e) {
                const int   lo = ((e & 7) >> 1) + 4 * khalf;      // j & 7
                const float r  = ((e < 8) ? rh0 : rh1) * rlo[lo];
                const int   p  = ((e < 8 ? pcom : pcom + 1) + __popc(lo)) & 3;
                float val;
                if (e & 1) val = (p == 1) ? -r : ((p == 3) ? r : 0.f);  // Im
                else       val = (p == 0) ?  r : ((p == 2) ? -r : 0.f); // Re
                af[e] = (_Float16)val;
            }

            // --- 8 WMMAs: this wave's 128 columns of the current half ---
#pragma unroll
            for (int t = 0; t < 8; ++t) {
                const int cl = cg * 128 + t * 16 + lsub;   // B col = lane%16
                const _Float16* bp = &Blds[cl][khalf * 8];
                v16h bf;
#pragma unroll
                for (int s = 0; s < 16; ++s) bf[s] = bp[(s & 7) + ((s >> 3) << 4)];
                acc[t] = __builtin_amdgcn_wmma_f32_16x16x32_f16(
                    false, af, false, bf, (short)0, acc[t], false, false);
            }
        }
        // signed |.|^2 : half 0 = cols(k<256) -> +1, half 1 -> -1
        const float s = half ? -1.f : 1.f;
#pragma unroll
        for (int t = 0; t < 8; ++t)
#pragma unroll
            for (int v = 0; v < 8; ++v) {
                const float d = acc[t][v];
                zacc[v] += s * d * d;
            }
    }

    // reduce over the 16 columns each lane-group holds (D: lanes 0-15 rows 0-7,
    // lanes 16-31 rows 8-15), then across the 4 col-group waves via LDS.
#pragma unroll
    for (int v = 0; v < 8; ++v) {
        float z = zacc[v];
        z += __shfl_xor(z, 1, 16);
        z += __shfl_xor(z, 2, 16);
        z += __shfl_xor(z, 4, 16);
        z += __shfl_xor(z, 8, 16);
        zacc[v] = z;
    }
    if (lsub == 0) {
#pragma unroll
        for (int v = 0; v < 8; ++v)
            red[cg][rg * 16 + khalf * 8 + v] = zacc[v];
    }
    __syncthreads();
    if (tid < 128)
        zrow[n0 + tid] = red[0][tid] + red[1][tid] + red[2][tid] + red[3][tid];
}

// ---------------------------------------------------------------------------
// Kernel 3: sum channels, broadcast over 16 output filters.
// ---------------------------------------------------------------------------
__global__ void finalize(const float* __restrict__ zrow, float* __restrict__ out) {
    const int o    = blockIdx.x * 256 + threadIdx.x;   // 65536 = 4*16*1024
    const int bimg = o >> 14;
    const int hw   = o & 1023;
    const float* zb = zrow + ((size_t)bimg << 12);
    out[o] = zb[hw] + zb[1024 + hw] + zb[2048 + hw] + zb[3072 + hw];
}

extern "C" void kernel_launch(void* const* d_in, const int* in_sizes, int n_in,
                              void* d_out, int out_size, void* d_ws, size_t ws_size,
                              hipStream_t stream) {
    (void)in_sizes; (void)n_in; (void)out_size; (void)ws_size;
    const float* x   = (const float*)d_in[0];   // (4,4,32,32) f32
    const float* wts = (const float*)d_in[1];   // (2,9) f32
    float*       out = (float*)d_out;           // (4,16,32,32) f32

    _Float16* Bt   = (_Float16*)d_ws;                                   // 2 MiB
    float*    zrow = (float*)((char*)d_ws + (size_t)1024 * 1024 * 2);   // 64 KiB

    build_U<<<512, 64, 0, stream>>>(wts, Bt);
    qgemm<<<128, 1024, 0, stream>>>(x, Bt, zrow);
    finalize<<<256, 256, 0, stream>>>(zrow, out);
}
